// ProteinWaterUpdate_66314295050606
// MI455X (gfx1250) — compile-verified, matrix operand order checked by reading, not run on hardware
//
#include <hip/hip_runtime.h>
#include <math.h>

// ---------------- problem dims ----------------
#define NP    4096
#define NW    12288
#define SD    512
#define VD    64
#define RBFD  16
#define KPW   12
#define KWW   8
#define NLAY  3
#define E_PW  (NP*KPW)     // 49152
#define E_WW  (NW*KWW)     // 98304
#define CHUNK 16384        // divides E_PW (3) and E_WW (6); NW=12288 fits in one chunk

typedef __bf16 bf16_t;
typedef __attribute__((ext_vector_type(16))) __bf16 v16bf;
typedef __attribute__((ext_vector_type(8)))  __bf16 v8bf;
typedef __attribute__((ext_vector_type(8)))  float  v8f;

#define CDIV(a,b) (((a)+(b)-1)/(b))

static const size_t PLANE_A  = (size_t)CHUNK * 160;  // vA / VhF / VhB component plane stride
static const size_t PLANE_B  = (size_t)CHUNK * 64;   // vB / Vo plane stride

// ================= WMMA bf16 GEMM =================
// C[M,N] f32 = A[M,Kp] bf16 (row major, lda) x Bt[N,Kp] bf16 (B transposed, ldb)
// M % 256 == 0, N % 32 == 0, K % 32 == 0.
// Each wave computes a 32x32 output (2x2 WMMA tiles); 8 waves stacked in M => 256x32 per block.
__device__ __forceinline__ v16bf load_frag(const bf16_t* base, int ld, int major0, int k0, int lane) {
  int m  = major0 + (lane & 15);
  int kb = k0 + ((lane >> 4) << 3);          // lanes 0-15: K k0..; lanes 16-31: K k0+8..
  const bf16_t* p = base + (size_t)m * ld + kb;
  v8bf lo = *(const v8bf*)(p);               // elems 0..7  -> K kb..kb+7
  v8bf hi = *(const v8bf*)(p + 16);          // elems 8..15 -> K kb+16..kb+23
  return __builtin_shufflevector(lo, hi, 0,1,2,3,4,5,6,7,8,9,10,11,12,13,14,15);
}

__global__ __launch_bounds__(256)
void gemm_bf16_wmma(const bf16_t* __restrict__ A, int lda,
                    const bf16_t* __restrict__ Bt, int ldb,
                    float* __restrict__ C, int ldc, int K) {
  int lane = threadIdx.x & 31;
  int wave = threadIdx.x >> 5;
  int n0 = blockIdx.x * 32;
  int m0 = blockIdx.y * 256 + wave * 32;
  v8f a00 = {}, a01 = {}, a10 = {}, a11 = {};
  for (int k = 0; k < K; k += 32) {
    v16bf fa0 = load_frag(A,  lda, m0,      k, lane);
    v16bf fa1 = load_frag(A,  lda, m0 + 16, k, lane);
    v16bf fb0 = load_frag(Bt, ldb, n0,      k, lane);
    v16bf fb1 = load_frag(Bt, ldb, n0 + 16, k, lane);
    a00 = __builtin_amdgcn_wmma_f32_16x16x32_bf16(false, fa0, false, fb0, (short)0, a00, false, false);
    a01 = __builtin_amdgcn_wmma_f32_16x16x32_bf16(false, fa0, false, fb1, (short)0, a01, false, false);
    a10 = __builtin_amdgcn_wmma_f32_16x16x32_bf16(false, fa1, false, fb0, (short)0, a10, false, false);
    a11 = __builtin_amdgcn_wmma_f32_16x16x32_bf16(false, fa1, false, fb1, (short)0, a11, false, false);
  }
  int col = lane & 15;
  int row = (lane >> 4) << 3;
  float* C00 = C + (size_t)(m0 + row) * ldc + n0 + col;
#pragma unroll
  for (int r = 0; r < 8; ++r) {
    C00[(size_t)r * ldc]                       = a00[r];   // rows m0+row..,   cols n0..
    C00[(size_t)r * ldc + 16]                  = a01[r];   // cols n0+16..
    C00[(size_t)(r + 16) * ldc]                = a10[r];   // rows m0+16+row..
    C00[(size_t)(r + 16) * ldc + 16]           = a11[r];
  }
}

// ================= weight convert: f32 [K,N] -> bf16 transposed padded [Np,Kp] =================
__global__ void conv_w_kernel(const float* __restrict__ W, bf16_t* __restrict__ Bt,
                              int K, int N, int Kp, int Np) {
  int i = blockIdx.x * 256 + threadIdx.x;
  if (i >= Np * Kp) return;
  int n = i / Kp, k = i % Kp;
  Bt[i] = (bf16_t)((k < K && n < N) ? W[(size_t)k * N + n] : 0.f);
}

// ================= brute-force KNN =================
template<int K, bool EXCL>
__global__ void knn_kernel(const float* __restrict__ posS, const float* __restrict__ posD,
                           int nD, int* __restrict__ out) {
  int s = blockIdx.x, tid = threadIdx.x;
  float sx = posS[3*s], sy = posS[3*s+1], sz = posS[3*s+2];
  float bd[K]; int bi[K];
#pragma unroll
  for (int i = 0; i < K; ++i) { bd[i] = 3e38f; bi[i] = 0; }
  for (int j = tid; j < nD; j += 256) {
    if (EXCL && j == s) continue;
    float dx = sx - posD[3*j], dy = sy - posD[3*j+1], dz = sz - posD[3*j+2];
    float d2 = dx*dx + dy*dy + dz*dz;
    if (d2 < bd[K-1]) {
      int p = K - 1;
      while (p > 0 && bd[p-1] > d2) { bd[p] = bd[p-1]; bi[p] = bi[p-1]; --p; }
      bd[p] = d2; bi[p] = j;
    }
  }
  __shared__ float rd[256]; __shared__ int ri[256]; __shared__ int ro[256];
  int head = 0;
  for (int r = 0; r < K; ++r) {
    rd[tid] = (head < K) ? bd[head] : 3e38f;
    ri[tid] = (head < K) ? bi[head] : 0;
    ro[tid] = tid;
    __syncthreads();
    for (int st = 128; st > 0; st >>= 1) {
      if (tid < st && rd[tid+st] < rd[tid]) { rd[tid]=rd[tid+st]; ri[tid]=ri[tid+st]; ro[tid]=ro[tid+st]; }
      __syncthreads();
    }
    if (tid == 0) out[s*K + r] = ri[0];
    int w = ro[0];
    __syncthreads();
    if (tid == w) ++head;
  }
}

// ================= edge gather: build [s_src|rbf|s_dst|0] bf16 and [v_src|unit|v_dst|0] bf16 =================
__global__ void msg_gather_kernel(const float* __restrict__ sS, const float* __restrict__ sD,
                                  const float* __restrict__ vS, const float* __restrict__ vD,
                                  const float* __restrict__ posS, const float* __restrict__ posD,
                                  const int* __restrict__ dstIdx, int kper, int base,
                                  bf16_t* __restrict__ sCat, int sld,
                                  bf16_t* __restrict__ vIn, size_t vPlane, int vld) {
  int e = blockIdx.x;
  int ge = base + e;
  int src = ge / kper;
  int dst = dstIdx[ge];
  float dx = posS[3*src]   - posD[3*dst];
  float dy = posS[3*src+1] - posD[3*dst+1];
  float dz = posS[3*src+2] - posD[3*dst+2];
  float d  = sqrtf(dx*dx + dy*dy + dz*dz + 1e-8f);
  float inv = 1.f / d;
  float unit[3] = { dx*inv, dy*inv, dz*inv };
  bf16_t* sr = sCat + (size_t)e * sld;
  for (int i = threadIdx.x; i < sld; i += 256) {
    float v;
    if (i < 512)       v = sS[(size_t)src*512 + i];
    else if (i < 528)  { float mu = (i-512) * (20.f/15.f); float t = (d-mu) * (1.f/1.25f); v = expf(-t*t); }
    else if (i < 1040) v = sD[(size_t)dst*512 + (i-528)];
    else               v = 0.f;
    sr[i] = (bf16_t)v;
  }
  for (int c = 0; c < 3; ++c) {
    bf16_t* vr = vIn + c*vPlane + (size_t)e * vld;
    for (int i = threadIdx.x; i < vld; i += 256) {
      float v;
      if (i < 64)       v = vS[((size_t)src*64 + i)*3 + c];
      else if (i == 64) v = unit[c];
      else if (i < 129) v = vD[((size_t)dst*64 + (i-65))*3 + c];
      else              v = 0.f;
      vr[i] = (bf16_t)v;
    }
  }
}

// ================= node gather for the update GVPs =================
__global__ void upd_gather_kernel(const float* __restrict__ sW, const float* __restrict__ ms,
                                  const float* __restrict__ vW, const float* __restrict__ mv,
                                  bf16_t* __restrict__ sCat, int sld,
                                  bf16_t* __restrict__ vIn, size_t vPlane, int vld) {
  int n = blockIdx.x;
  bf16_t* sr = sCat + (size_t)n * sld;
  for (int i = threadIdx.x; i < sld; i += 256) {
    float v = (i < 512) ? sW[(size_t)n*512 + i] : (i < 1024) ? ms[(size_t)n*512 + (i-512)] : 0.f;
    sr[i] = (bf16_t)v;
  }
  for (int c = 0; c < 3; ++c) {
    bf16_t* vr = vIn + c*vPlane + (size_t)n * vld;
    for (int i = threadIdx.x; i < vld; i += 256) {
      float v = (i < 64) ? vW[((size_t)n*64 + i)*3 + c] : mv[((size_t)n*64 + (i-64))*3 + c];
      vr[i] = (bf16_t)v;
    }
  }
}

// ================= vector norms; also bf16 copy of Vh =================
__global__ void vn_kernel(const float* __restrict__ Vh, size_t plane, int hp, int h,
                          bf16_t* __restrict__ Vhb, bf16_t* __restrict__ sCat, int sld, int si,
                          int total /* M*hp */) {
  int i = blockIdx.x * 256 + threadIdx.x;
  if (i >= total) return;
  int m = i / hp, col = i % hp;
  float x = Vh[i], y = Vh[plane + i], z = Vh[2*plane + i];
  Vhb[i] = (bf16_t)x; Vhb[plane + i] = (bf16_t)y; Vhb[2*plane + i] = (bf16_t)z;
  if (col < h)
    sCat[(size_t)m * sld + si + col] = (bf16_t)sqrtf(x*x + y*y + z*z + 1e-8f);
}

// ================= bias + (relu) ; also bf16 copy for next stage =================
__global__ void act_kernel(float* __restrict__ sPre, const float* __restrict__ bias, int relu,
                           bf16_t* __restrict__ sOut, int old, int total /* M*512 */) {
  int i = blockIdx.x * 256 + threadIdx.x;
  if (i >= total) return;
  int col = i & 511;
  float v = sPre[i] + bias[col];
  if (relu) v = fmaxf(v, 0.f);
  sPre[i] = v;
  sOut[(size_t)(i >> 9) * old + col] = (bf16_t)v;
}

// ================= sigmoid vector gating; also bf16 copy =================
__global__ void gate_kernel(const float* __restrict__ gPre, float* __restrict__ Vo, size_t voPlane,
                            bf16_t* __restrict__ vOut, size_t vOutPlane, int total /* M*64 */) {
  int i = blockIdx.x * 256 + threadIdx.x;
  if (i >= total) return;
  float gv = 1.f / (1.f + expf(-gPre[i]));
#pragma unroll
  for (int c = 0; c < 3; ++c) {
    float v = Vo[c*voPlane + i] * gv;
    Vo[c*voPlane + i] = v;
    vOut[c*vOutPlane + i] = (bf16_t)v;
  }
}

// ================= segment-sum scatters =================
__global__ void scatter_s_kernel(const float* __restrict__ sOut, const int* __restrict__ dstIdx,
                                 int base, float* __restrict__ ms, int total /* M*512 */) {
  int i = blockIdx.x * 256 + threadIdx.x;
  if (i >= total) return;
  int m = i >> 9, col = i & 511;
  int dst = dstIdx[base + m];
  atomicAdd(&ms[(size_t)dst * 512 + col], sOut[i]);
}

__global__ void scatter_v_kernel(const float* __restrict__ Vo, size_t plane,
                                 const int* __restrict__ dstIdx, int base,
                                 float* __restrict__ mv, int total /* M*192 */) {
  int i = blockIdx.x * 256 + threadIdx.x;
  if (i >= total) return;
  int m = i / 192, r = i % 192;
  int j = r / 3, c = r % 3;
  int dst = dstIdx[base + m];
  atomicAdd(&mv[((size_t)dst * 64 + j) * 3 + c], Vo[c*plane + (size_t)m*64 + j]);
}

// ================= residual =================
__global__ void residual_kernel(float* __restrict__ sW, float* __restrict__ vW,
                                const float* __restrict__ sOut, const float* __restrict__ Vo,
                                size_t voPlane) {
  int i = blockIdx.x * 256 + threadIdx.x;
  const int sTot = NW * 512;
  const int tot = sTot + NW * 192;
  if (i >= tot) return;
  if (i < sTot) { sW[i] += sOut[i]; }
  else {
    int j = i - sTot; int c = j % 3; int t = j / 3;     // t = n*64 + v
    vW[j] += Vo[c*voPlane + t];
  }
}

// ================= host orchestration =================
struct GD { int si, vi, h, vip, hp, sihp; };   // so=512, vo=64 fixed

extern "C" void kernel_launch(void* const* d_in, const int* in_sizes, int n_in,
                              void* d_out, int out_size, void* d_ws, size_t ws_size,
                              hipStream_t stream) {
  const float* s_p   = (const float*)d_in[0];
  const float* v_p   = (const float*)d_in[1];
  const float* s_w0  = (const float*)d_in[2];
  const float* v_w0  = (const float*)d_in[3];
  const float* pos_p = (const float*)d_in[4];
  const float* pos_w = (const float*)d_in[5];
  // params flattened in dict insertion order: layer b { pw g0-2, ww g0-2, upd g0-2 },
  // each GVP { Wh, Wv, Ws, bs, Wg }
  auto P = [&](int b, int et, int g, int f) -> const float* {
    return (const float*)d_in[6 + (((b*3 + et)*3 + g)*5 + f)];
  };

  static const GD MSG_D[3] = {{1040,129,129,160,160,1184},{512,64,64,64,64,576},{512,64,64,64,64,576}};
  static const GD UPD_D[3] = {{1024,128,128,128,128,1152},{512,64,64,64,64,576},{512,64,64,64,64,576}};

  // ---- workspace bump allocator ----
  char* wsb = (char*)d_ws;
  size_t off = 0;
  auto alloc = [&](size_t bytes) -> void* {
    off = (off + 255) & ~(size_t)255;
    void* p = wsb + off;
    off += bytes;
    return p;
  };

  int*    idx_pw  = (int*)   alloc((size_t)E_PW * 4);
  int*    idx_ww  = (int*)   alloc((size_t)E_WW * 4);
  float*  s_w_cur = (float*) alloc((size_t)NW * SD * 4);
  float*  v_w_cur = (float*) alloc((size_t)NW * VD * 3 * 4);
  float*  msBuf   = (float*) alloc((size_t)NW * SD * 4);
  float*  mvBuf   = (float*) alloc((size_t)NW * VD * 3 * 4);
  bf16_t* sA      = (bf16_t*)alloc((size_t)CHUNK * 1184 * 2);
  bf16_t* sB      = (bf16_t*)alloc((size_t)CHUNK * 576 * 2);
  bf16_t* vA      = (bf16_t*)alloc(3 * PLANE_A * 2);
  bf16_t* vB      = (bf16_t*)alloc(3 * PLANE_B * 2);
  float*  VhF     = (float*) alloc(3 * PLANE_A * 4);
  bf16_t* VhB     = (bf16_t*)alloc(3 * PLANE_A * 2);
  float*  sPre    = (float*) alloc((size_t)CHUNK * 512 * 4);
  float*  gPre    = (float*) alloc((size_t)CHUNK * 64 * 4);
  float*  VoF     = (float*) alloc(3 * PLANE_B * 4);

  // ---- convert all weights: bf16, transposed [Np, Kp], zero padded ----
  bf16_t* Wht[27]; bf16_t* Wst[27]; bf16_t* Wvt[27]; bf16_t* Wgt[27];
  for (int b = 0; b < NLAY; ++b)
    for (int et = 0; et < 3; ++et)
      for (int g = 0; g < 3; ++g) {
        int p = (b*3 + et)*3 + g;
        const GD& D = (et == 2) ? UPD_D[g] : MSG_D[g];
        Wht[p] = (bf16_t*)alloc((size_t)D.hp * D.vip * 2);
        Wst[p] = (bf16_t*)alloc((size_t)512 * D.sihp * 2);
        Wvt[p] = (bf16_t*)alloc((size_t)64 * D.hp * 2);
        Wgt[p] = (bf16_t*)alloc((size_t)64 * 512 * 2);
        conv_w_kernel<<<CDIV(D.hp*D.vip,256),256,0,stream>>>(P(b,et,g,0), Wht[p], D.vi, D.h, D.vip, D.hp);
        conv_w_kernel<<<CDIV(512*D.sihp,256),256,0,stream>>>(P(b,et,g,2), Wst[p], D.si+D.h, 512, D.sihp, 512);
        conv_w_kernel<<<CDIV(64*D.hp,256),256,0,stream>>>(P(b,et,g,1), Wvt[p], D.h, 64, D.hp, 64);
        conv_w_kernel<<<CDIV(64*512,256),256,0,stream>>>(P(b,et,g,4), Wgt[p], 512, 64, 512, 64);
      }

  // ---- KNN (layer independent) ----
  knn_kernel<KPW,false><<<NP,256,0,stream>>>(pos_p, pos_w, NW, idx_pw);
  knn_kernel<KWW,true ><<<NW,256,0,stream>>>(pos_w, pos_w, NW, idx_ww);

  // ---- init node state ----
  hipMemcpyAsync(s_w_cur, s_w0, (size_t)NW*SD*4,   hipMemcpyDeviceToDevice, stream);
  hipMemcpyAsync(v_w_cur, v_w0, (size_t)NW*VD*3*4, hipMemcpyDeviceToDevice, stream);

  auto gemm = [&](const bf16_t* A, int lda, const bf16_t* Bt, int ldb,
                  float* C, int ldc, int M, int N, int K) {
    dim3 grid(N/32, M/256);
    gemm_bf16_wmma<<<grid, 256, 0, stream>>>(A, lda, Bt, ldb, C, ldc, K);
  };

  // 3-GVP chain. Inputs start in sA (ld = sld0) / vA (plane PLANE_A, ld = dims[0].vip).
  // Final f32 outputs: sPre [M,512], VoF [3][M,64].
  auto chain = [&](const GD* dims, int b, int et, int M, int sld0) {
    bf16_t* sBufs[2]  = { sA, sB };
    bf16_t* vBufs[2]  = { vA, vB };
    size_t  vPlanes[2] = { PLANE_A, PLANE_B };
    int cur = 0;
    int sld = sld0;
    int vld = dims[0].vip;
    for (int g = 0; g < 3; ++g) {
      int p = (b*3 + et)*3 + g;
      const GD& D = dims[g];
      int nxt = cur ^ 1;
      // Vh = V x Wh  (per component)
      for (int c = 0; c < 3; ++c)
        gemm(vBufs[cur] + c*vPlanes[cur], vld, Wht[p], D.vip,
             VhF + c*PLANE_A, D.hp, M, D.hp, D.vip);
      // vector norms -> tail of scalar concat; Vh -> bf16
      vn_kernel<<<CDIV(M*D.hp,256),256,0,stream>>>(VhF, PLANE_A, D.hp, D.h,
                                                   VhB, sBufs[cur], sld, D.si, M*D.hp);
      // s_pre = [s|vn] x Ws
      gemm(sBufs[cur], sld, Wst[p], D.sihp, sPre, 512, M, 512, D.sihp);
      // bias + relu (not on last GVP); bf16 copy to next scalar buffer
      act_kernel<<<CDIV(M*512,256),256,0,stream>>>(sPre, P(b,et,g,3), (g < 2) ? 1 : 0,
                                                   sBufs[nxt], 576, M*512);
      // gate pre-activation = s_out x Wg
      gemm(sBufs[nxt], 576, Wgt[p], 512, gPre, 64, M, 64, 512);
      // Vo = Vh x Wv (per component)
      for (int c = 0; c < 3; ++c)
        gemm(VhB + c*PLANE_A, D.hp, Wvt[p], D.hp, VoF + c*PLANE_B, 64, M, 64, D.hp);
      // Vo *= sigmoid(gate); bf16 copy to next vector buffer
      gate_kernel<<<CDIV(M*64,256),256,0,stream>>>(gPre, VoF, PLANE_B,
                                                   vBufs[nxt], vPlanes[nxt], M*64);
      cur = nxt; sld = 576; vld = 64;
    }
  };

  auto run_msg = [&](int b, int et, const float* sS, const float* vS, const float* posS,
                     int nE, int kper, const int* dstIdx) {
    for (int base = 0; base < nE; base += CHUNK) {
      msg_gather_kernel<<<CHUNK,256,0,stream>>>(sS, s_w_cur, vS, v_w_cur, posS, pos_w,
                                                dstIdx, kper, base,
                                                sA, 1184, vA, PLANE_A, 160);
      chain(MSG_D, b, et, CHUNK, 1184);
      scatter_s_kernel<<<CDIV(CHUNK*512,256),256,0,stream>>>(sPre, dstIdx, base, msBuf, CHUNK*512);
      scatter_v_kernel<<<CDIV(CHUNK*192,256),256,0,stream>>>(VoF, PLANE_B, dstIdx, base, mvBuf, CHUNK*192);
    }
  };

  for (int b = 0; b < NLAY; ++b) {
    hipMemsetAsync(msBuf, 0, (size_t)NW*SD*4,   stream);
    hipMemsetAsync(mvBuf, 0, (size_t)NW*VD*3*4, stream);
    run_msg(b, 0, s_p,     v_p,     pos_p, E_PW, KPW, idx_pw);   // protein -> water
    run_msg(b, 1, s_w_cur, v_w_cur, pos_w, E_WW, KWW, idx_ww);   // water -> water
    // update GVPs on [node | aggregated messages]
    upd_gather_kernel<<<NW,256,0,stream>>>(s_w_cur, msBuf, v_w_cur, mvBuf,
                                           sA, 1152, vA, PLANE_A, 128);
    chain(UPD_D, b, 2, NW, 1152);
    residual_kernel<<<CDIV(NW*512+NW*192,256),256,0,stream>>>(s_w_cur, v_w_cur, sPre, VoF, PLANE_B);
  }

  // ---- outputs: (s_p, v_p, s_w, v_w) concatenated flat, f32 ----
  float* out = (float*)d_out;
  size_t o0 = 0;
  size_t o1 = o0 + (size_t)NP*SD;
  size_t o2 = o1 + (size_t)NP*VD*3;
  size_t o3 = o2 + (size_t)NW*SD;
  hipMemcpyAsync(out + o0, s_p,     (size_t)NP*SD*4,   hipMemcpyDeviceToDevice, stream);
  hipMemcpyAsync(out + o1, v_p,     (size_t)NP*VD*3*4, hipMemcpyDeviceToDevice, stream);
  hipMemcpyAsync(out + o2, s_w_cur, (size_t)NW*SD*4,   hipMemcpyDeviceToDevice, stream);
  hipMemcpyAsync(out + o3, v_w_cur, (size_t)NW*VD*3*4, hipMemcpyDeviceToDevice, stream);
  (void)in_sizes; (void)n_in; (void)out_size; (void)ws_size;
}
